// GRUNetwork_44470091383191
// MI455X (gfx1250) — matrix operand order repaired and checked by reading, not compiled, not run on hardware
//
#include <hip/hip_runtime.h>
#include <hip/hip_bf16.h>

// GRU: B=64, T=2048, D=H=O=1024. bf16 WMMA (v_wmma_f32_16x16x32_bf16) for all
// GEMMs; persistent grid-barrier kernel for the sequential T-scan (compiler
// keeps the full Wh weight block resident in 768 high VGPRs via vgpr-msb).

#define GB 64
#define GT 2048
#define GD 1024
#define GH 1024
#define GO 1024
#define G3H 3072
#define BQ (GB * GH)        // 65536 elements per h buffer
#define SCAN_WGS 64

typedef __attribute__((ext_vector_type(16))) __bf16 v16bf;
typedef __attribute__((ext_vector_type(8)))  float  v8f;

// ---------- helpers ----------
__device__ __forceinline__ unsigned short f2bf(float f) {
    unsigned u = __float_as_uint(f);
    unsigned r = 0x7FFFu + ((u >> 16) & 1u);      // round-to-nearest-even
    return (unsigned short)((u + r) >> 16);
}
__device__ __forceinline__ float bf2f(unsigned short h) {
    return __uint_as_float(((unsigned)h) << 16);
}
__device__ __forceinline__ float sigmoidf_(float x) {
    return 1.0f / (1.0f + __expf(-x));
}

union FragU { v16bf v; uint4 q[2]; };

// A fragment (16x32 bf16, MxK): lanes 0-15 -> M=lane, K 0-7 & 16-23;
// lanes 16-31 -> M=lane-16, K 8-15 & 24-31. Two 16B contiguous chunks.
__device__ __forceinline__ v16bf load_a_frag(const unsigned short* A, int lda,
                                             int mo, int ko, int lane) {
    int half = lane >> 4, m = lane & 15;
    const unsigned short* p = A + (size_t)(mo + m) * lda + ko + 8 * half;
    FragU f;
    f.q[0] = *(const uint4*)(const void*)p;        // K = 8*half .. +7
    f.q[1] = *(const uint4*)(const void*)(p + 16); // K = 16+8*half .. +7
    return f.v;
}

// B fragment (32x16 bf16, KxN) from row-major W[n, k] (i.e. W^T columns):
// lanes 0-15 -> N=lane, K 0-7 & 16-23; lanes 16-31 -> N=lane-16, K 8-15 & 24-31.
__device__ __forceinline__ v16bf load_b_frag(const unsigned short* W, int ldw,
                                             int no, int ko, int lane) {
    int half = lane >> 4, n = lane & 15;
    const unsigned short* p = W + (size_t)(no + n) * ldw + ko + 8 * half;
    FragU f;
    f.q[0] = *(const uint4*)(const void*)p;        // K = 8*half .. +7
    f.q[1] = *(const uint4*)(const void*)(p + 16); // K = 16+8*half .. +7
    return f.v;
}

__device__ __forceinline__ v8f wmma_bf16(v16bf a, v16bf b, v8f c) {
    return __builtin_amdgcn_wmma_f32_16x16x32_bf16(
        false, a, false, b, (short)0, c, false, false);
}

// ---------- fp32 -> bf16 conversion (vectorized, n multiple of 4) ----------
__global__ void __launch_bounds__(256) cvt_bf16x4(const float4* __restrict__ in,
                                                  uint2* __restrict__ out,
                                                  size_t n4) {
    size_t i = (size_t)blockIdx.x * blockDim.x + threadIdx.x;
    size_t stride = (size_t)gridDim.x * blockDim.x;
    for (; i < n4; i += stride) {
        float4 v = in[i];
        uint2 o;
        o.x = (unsigned)f2bf(v.x) | ((unsigned)f2bf(v.y) << 16);
        o.y = (unsigned)f2bf(v.z) | ((unsigned)f2bf(v.w) << 16);
        out[i] = o;
    }
}

// ---------- zero h state + barrier every call ----------
__global__ void __launch_bounds__(256) init_state(float* __restrict__ hF,
                                                  unsigned short* __restrict__ hB,
                                                  unsigned* __restrict__ bar) {
    size_t i = (size_t)blockIdx.x * blockDim.x + threadIdx.x;
    if (i < (size_t)2 * BQ) { hF[i] = 0.0f; hB[i] = 0; }
    if (i < 2) bar[i] = 0u;
}

// ---------- generic WMMA GEMM: out[M,N] = A[M,K](bf16) @ W[N,K]^T + bias ----
// WG = 256 threads (8 waves) arranged 2x4; wave tile = (MI*16) x (NJ*16)
// subtiles, so WG tile = (2*MI*16) x (4*NJ*16).
// MI=NJ=4: wave 64x64, WG 128x256  -> 16 WMMA per 8 b128-frag loads (~32 f/B)
// MI=NJ=2: wave 32x32, WG  64x128  -> small-M fallback for the logits GEMM
template <int MI, int NJ, bool OUT_BF16>
__global__ void __launch_bounds__(256) wmma_gemm_bias(
    const unsigned short* __restrict__ A, int lda,
    const unsigned short* __restrict__ W, int ldw,
    const float* __restrict__ bias,
    void* __restrict__ out, int ldo, int K) {
    const int lane = threadIdx.x & 31;
    const int w = threadIdx.x >> 5;
    const int m0 = blockIdx.x * (2 * MI * 16) + (w & 1) * (MI * 16);
    const int n0 = blockIdx.y * (4 * NJ * 16) + (w >> 1) * (NJ * 16);

    v8f acc[MI][NJ] = {};
    for (int ko = 0; ko < K; ko += 32) {
        v16bf fa[MI], fb[NJ];
#pragma unroll
        for (int i = 0; i < MI; ++i)
            fa[i] = load_a_frag(A, lda, m0 + i * 16, ko, lane);
#pragma unroll
        for (int j = 0; j < NJ; ++j)
            fb[j] = load_b_frag(W, ldw, n0 + j * 16, ko, lane);
#pragma unroll
        for (int i = 0; i < MI; ++i)
#pragma unroll
            for (int j = 0; j < NJ; ++j)
                acc[i][j] = wmma_bf16(fa[i], fb[j], acc[i][j]);
    }

    const int half = lane >> 4, nl = lane & 15;
#pragma unroll
    for (int i = 0; i < MI; ++i)
#pragma unroll
        for (int j = 0; j < NJ; ++j) {
            const int col = n0 + j * 16 + nl;
            const float bv = bias[col];
#pragma unroll
            for (int r = 0; r < 8; ++r) {
                const int row = m0 + i * 16 + half * 8 + r;
                const float v = acc[i][j][r] + bv;
                if (OUT_BF16)
                    ((unsigned short*)out)[(size_t)row * ldo + col] = f2bf(v);
                else
                    ((float*)out)[(size_t)row * ldo + col] = v;
            }
        }
}

// ---------- sense-reversing grid barrier ----------
__device__ __forceinline__ void grid_barrier(unsigned* cnt, unsigned* gen,
                                             unsigned nwg) {
    __syncthreads();
    if (threadIdx.x == 0) {
        __threadfence();
        unsigned g = ((volatile unsigned*)gen)[0];
        if (atomicAdd(cnt, 1u) == nwg - 1u) {
            ((volatile unsigned*)cnt)[0] = 0u;
            __threadfence();
            atomicAdd(gen, 1u);
        } else {
            while (((volatile unsigned*)gen)[0] == g)
                __builtin_amdgcn_s_sleep(2);
        }
        __threadfence();
    }
    __syncthreads();
}

// ---------- persistent GRU scan ----------
// Grid: 64 WGs x 128 threads. WG -> (batch tile bt in 0..3) x (col tile ct in
// 0..15, 64 cols). Wave w handles 16 columns; computes r/u/n gate GEMMs
// (shared A fragment, 3 WMMA accumulator sets), then gating + h update.
// Codegen keeps all 96 Wh fragments (768 VGPRs) live across the t-loop.
__global__ void __launch_bounds__(128) gru_scan(
    const unsigned short* __restrict__ xp,    // [B*T, 3H] bf16 (= x@Wx^T + bx)
    const unsigned short* __restrict__ whbf,  // [3H, H] bf16
    const float* __restrict__ bh,             // [3H]
    float* __restrict__ hF,                   // [2][B*H] fp32
    unsigned short* __restrict__ hB,          // [2][B*H] bf16
    unsigned* __restrict__ bar) {
    const int lane = threadIdx.x & 31;
    const int w = threadIdx.x >> 5;
    const int bt = blockIdx.x & 3;
    const int ct = blockIdx.x >> 2;
    const int m0 = bt * 16;
    const int n0 = ct * 64 + w * 16;
    const int half = lane >> 4, nl = lane & 15;
    const int col = n0 + nl;
    const float bhr = bh[col], bhu = bh[GH + col], bhn = bh[2 * GH + col];

    for (int t = 0; t < GT; ++t) {
        const int cur = t & 1, nxt = cur ^ 1;
        const unsigned short* hcur = hB + (size_t)cur * BQ;

        v8f ar = {}, au = {}, an = {};
        for (int ko = 0; ko < GH; ko += 32) {
            v16bf fa = load_a_frag(hcur, GH, m0, ko, lane);
            v16bf br = load_b_frag(whbf, GH, n0, ko, lane);
            v16bf bu = load_b_frag(whbf, GH, GH + n0, ko, lane);
            v16bf bn = load_b_frag(whbf, GH, 2 * GH + n0, ko, lane);
            ar = wmma_bf16(fa, br, ar);
            au = wmma_bf16(fa, bu, au);
            an = wmma_bf16(fa, bn, an);
        }

        for (int r = 0; r < 8; ++r) {
            const int row = m0 + half * 8 + r;                 // batch index
            const size_t xbase = ((size_t)row * GT + t) * G3H;
            const float xr = bf2f(xp[xbase + col]);
            const float xu = bf2f(xp[xbase + GH + col]);
            const float xn = bf2f(xp[xbase + 2 * GH + col]);
            const float rg = sigmoidf_(xr + ar[r] + bhr);
            const float ug = sigmoidf_(xu + au[r] + bhu);
            const float ng = tanhf(xn + rg * (an[r] + bhn));
            const size_t hidx = (size_t)row * GH + col;
            const float hprev = hF[(size_t)cur * BQ + hidx];
            const float hnew = ug * hprev + (1.0f - ug) * ng;
            hF[(size_t)nxt * BQ + hidx] = hnew;
            hB[(size_t)nxt * BQ + hidx] = f2bf(hnew);
            if (t + 1 < GT)  // warm L2 for next step's x_proj slice
                __builtin_prefetch(&xp[xbase + G3H + col], 0, 1);
        }
        grid_barrier(&bar[0], &bar[1], (unsigned)gridDim.x);
    }
}

// ---------- per-row log-softmax ----------
__global__ void __launch_bounds__(256) logsoftmax_rows(
    const float* __restrict__ logits, float* __restrict__ out) {
    __shared__ float red[256];
    const int row = blockIdx.x;
    const float* p = logits + (size_t)row * GO;
    float mx = -INFINITY;
    for (int c = threadIdx.x; c < GO; c += 256) mx = fmaxf(mx, p[c]);
    red[threadIdx.x] = mx;
    __syncthreads();
    for (int s = 128; s > 0; s >>= 1) {
        if (threadIdx.x < s) red[threadIdx.x] = fmaxf(red[threadIdx.x], red[threadIdx.x + s]);
        __syncthreads();
    }
    mx = red[0];
    __syncthreads();
    float sum = 0.0f;
    for (int c = threadIdx.x; c < GO; c += 256) sum += __expf(p[c] - mx);
    red[threadIdx.x] = sum;
    __syncthreads();
    for (int s = 128; s > 0; s >>= 1) {
        if (threadIdx.x < s) red[threadIdx.x] += red[threadIdx.x + s];
        __syncthreads();
    }
    const float lse = mx + __logf(red[0]);
    for (int c = threadIdx.x; c < GO; c += 256)
        out[(size_t)row * GO + c] = p[c] - lse;
}

// ---------- host ----------
extern "C" void kernel_launch(void* const* d_in, const int* in_sizes, int n_in,
                              void* d_out, int out_size, void* d_ws, size_t ws_size,
                              hipStream_t stream) {
    (void)in_sizes; (void)n_in; (void)out_size; (void)ws_size;
    const float* x  = (const float*)d_in[0];   // [B,T,D]
    const float* Wx = (const float*)d_in[1];   // [3H,D]
    const float* bx = (const float*)d_in[2];   // [3H]
    const float* Wh = (const float*)d_in[3];   // [3H,H]
    const float* bh = (const float*)d_in[4];   // [3H]
    const float* Wf = (const float*)d_in[5];   // [O,H]
    const float* bf = (const float*)d_in[6];   // [O]
    float* outp = (float*)d_out;               // [B,O] fp32

    char* ws = (char*)d_ws;
    size_t off = 0;
    auto alloc = [&](size_t bytes) -> char* {
        char* p = ws + off;
        off += (bytes + 255) & ~(size_t)255;
        return p;
    };
    const size_t nX  = (size_t)GB * GT * GD;        // 134,217,728
    const size_t nWx = (size_t)G3H * GD;            // 3,145,728
    const size_t nWh = (size_t)G3H * GH;
    const size_t nWf = (size_t)GO * GH;             // 1,048,576
    const size_t nXP = (size_t)GB * GT * G3H;       // 402,653,184

    unsigned short* xbf   = (unsigned short*)alloc(nX * 2);
    unsigned short* wxbf  = (unsigned short*)alloc(nWx * 2);
    unsigned short* whbf  = (unsigned short*)alloc(nWh * 2);
    unsigned short* wfbf  = (unsigned short*)alloc(nWf * 2);
    unsigned short* xpbf  = (unsigned short*)alloc(nXP * 2);
    float*          hF    = (float*)alloc((size_t)2 * BQ * 4);
    unsigned short* hB    = (unsigned short*)alloc((size_t)2 * BQ * 2);
    float*          logit = (float*)alloc((size_t)GB * GO * 4);
    unsigned*       bar   = (unsigned*)alloc(256);

    // 1) fp32 -> bf16 conversions (streaming, ~0.8 GB total traffic)
    cvt_bf16x4<<<8192, 256, 0, stream>>>((const float4*)x,  (uint2*)xbf,  nX  / 4);
    cvt_bf16x4<<<3072, 256, 0, stream>>>((const float4*)Wx, (uint2*)wxbf, nWx / 4);
    cvt_bf16x4<<<3072, 256, 0, stream>>>((const float4*)Wh, (uint2*)whbf, nWh / 4);
    cvt_bf16x4<<<1024, 256, 0, stream>>>((const float4*)Wf, (uint2*)wfbf, nWf / 4);

    // 2) zero h0 (both buffers) + barrier state (ws is poisoned between calls)
    init_state<<<512, 256, 0, stream>>>(hF, hB, bar);

    // 3) x_proj = x @ Wx^T + bx   (bf16 out, 825 GFLOP of WMMA; 128x256 tiles)
    wmma_gemm_bias<4, 4, true>
        <<<dim3((GB * GT) / 128, G3H / 256), 256, 0, stream>>>(
            xbf, GD, wxbf, GD, bx, (void*)xpbf, G3H, GD);

    // 4) persistent sequential scan over T with grid barrier per step
    gru_scan<<<SCAN_WGS, 128, 0, stream>>>(xpbf, whbf, bh, hF, hB, bar);

    // 5) logits = h_last @ Wf^T + bf   (h_last ends in buffer 0: T even)
    wmma_gemm_bias<2, 2, false>
        <<<dim3(GB / 64, GO / 128), 256, 0, stream>>>(
            hB, GH, wfbf, GH, bf, (void*)logit, GO, GH);

    // 6) row-wise log-softmax -> d_out
    logsoftmax_rows<<<GB, 256, 0, stream>>>(logit, outp);
}
